// D_Attention_21303037788405
// MI455X (gfx1250) — compile-verified, matrix operand order checked by reading, not run on hardware
//
#include <hip/hip_runtime.h>
#include <hip/hip_bf16.h>

typedef unsigned short u16;
typedef unsigned int   u32;
typedef __attribute__((ext_vector_type(16))) __bf16 v16bf;
typedef __attribute__((ext_vector_type(8)))  float  v8f;

// ---------- helpers ----------
__device__ __forceinline__ u16 f2bf(float x) {          // round-half-up, 2 VALU
    return (u16)((__float_as_uint(x) + 0x8000u) >> 16);
}
// two f32 -> packed 2xbf16 with one v_perm_b32 (3 VALU per 2 elements)
__device__ __forceinline__ u32 pack2bf(float lo, float hi) {
    return __builtin_amdgcn_perm(__float_as_uint(hi) + 0x8000u,
                                 __float_as_uint(lo) + 0x8000u, 0x07060302u);
}
__device__ __forceinline__ void st8bf(u16* dst, float4 r0, float4 r1) {
    uint4 q;
    q.x = pack2bf(r0.x, r0.y); q.y = pack2bf(r0.z, r0.w);
    q.z = pack2bf(r1.x, r1.y); q.w = pack2bf(r1.z, r1.w);
    *reinterpret_cast<uint4*>(dst) = q;
}
__device__ __forceinline__ v16bf load_frag(const u16* rowp, int kbase) {
    union { uint4 q[2]; v16bf v; } u;
    u.q[0] = *reinterpret_cast<const uint4*>(rowp + kbase);
    u.q[1] = *reinterpret_cast<const uint4*>(rowp + kbase + 16);
    return u.v;
}
__device__ __forceinline__ v8f bwmma(v16bf a, v16bf b, v8f c) {
    return __builtin_amdgcn_wmma_f32_16x16x32_bf16(false, a, false, b, (short)0, c, false, false);
}

#define LDS_PITCH 40   // u16 units: 80 B/row -> conflict-free b128 reads

// ---------- gemm_big: 128x64 tile, 32x32/wave (4 WMMA / 4 frag loads) ----------
// C[M,N] = A[M,K]*B[N,K]^T + bias, opt ReLU. Edge-guarded on M and K tail.
__global__ __launch_bounds__(256) void gemm_big(
    const float* __restrict__ A, const float* __restrict__ B,
    float* __restrict__ C, const float* __restrict__ bias,
    int M, int N, int K, int act)
{
    __shared__ alignas(16) u16 sA[2][128 * LDS_PITCH];
    __shared__ alignas(16) u16 sB[2][64 * LDS_PITCH];

    const int tid  = threadIdx.x;
    const int lane = tid & 31;
    const int wave = tid >> 5;
    const int wm   = wave >> 1;   // 0..3 : 32-row slab
    const int wn   = wave & 1;    // 0..1 : 32-col slab
    const int m0   = blockIdx.y * 128;
    const int n0   = blockIdx.x * 64;

    const int  srow  = tid >> 2;        // 0..63
    const int  skof  = (tid & 3) * 8;
    const bool edgeM = (m0 + 128 > M);  // block-uniform
    const int  gm0 = m0 + srow, gm1 = m0 + 64 + srow, gn = n0 + srow;
    const float* aRow0 = A + (size_t)gm0 * K + skof;
    const float* aRow1 = A + (size_t)gm1 * K + skof;
    const float* bRow  = B + (size_t)gn  * K + skof;

    v8f acc00 = {}, acc01 = {}, acc10 = {}, acc11 = {};
    const int nk = (K + 31) >> 5;
    float4 a00, a01, a10, a11, bb0, bb1;

    auto load_chunk = [&](int kc) {
        if (!edgeM && (kc + 32 <= K)) {
            const float4* p0 = reinterpret_cast<const float4*>(aRow0 + kc);
            const float4* p1 = reinterpret_cast<const float4*>(aRow1 + kc);
            const float4* pb = reinterpret_cast<const float4*>(bRow + kc);
            a00 = p0[0]; a01 = p0[1]; a10 = p1[0]; a11 = p1[1];
            bb0 = pb[0]; bb1 = pb[1];
        } else {
            float t0[8], t1[8], tb[8];
            #pragma unroll
            for (int j = 0; j < 8; ++j) {
                const int kk = kc + skof + j;
                const bool ok = kk < K;
                t0[j] = (ok && gm0 < M) ? aRow0[kc + j] : 0.0f;
                t1[j] = (ok && gm1 < M) ? aRow1[kc + j] : 0.0f;
                tb[j] = (ok && gn  < N) ? bRow [kc + j] : 0.0f;
            }
            a00 = make_float4(t0[0], t0[1], t0[2], t0[3]); a01 = make_float4(t0[4], t0[5], t0[6], t0[7]);
            a10 = make_float4(t1[0], t1[1], t1[2], t1[3]); a11 = make_float4(t1[4], t1[5], t1[6], t1[7]);
            bb0 = make_float4(tb[0], tb[1], tb[2], tb[3]); bb1 = make_float4(tb[4], tb[5], tb[6], tb[7]);
        }
    };
    auto store_chunk = [&](int buf) {
        st8bf(sA[buf] + srow * LDS_PITCH + skof, a00, a01);
        st8bf(sA[buf] + (64 + srow) * LDS_PITCH + skof, a10, a11);
        st8bf(sB[buf] + srow * LDS_PITCH + skof, bb0, bb1);
    };

    load_chunk(0);
    store_chunk(0);
    __syncthreads();

    const int r  = lane & 15;
    const int kb = (lane >> 4) * 8;
    const size_t aStride = 128 * LDS_PITCH, bStride = 64 * LDS_PITCH;
    const u16* fA0 = sA[0] + (wm * 32 + r) * LDS_PITCH;
    const u16* fA1 = fA0 + 16 * LDS_PITCH;
    const u16* fB0 = sB[0] + (wn * 32 + r) * LDS_PITCH;
    const u16* fB1 = fB0 + 16 * LDS_PITCH;

    for (int i = 0; i < nk; ++i) {
        const int cur = i & 1;
        const bool more = (i + 1 < nk);
        if (more) load_chunk((i + 1) << 5);          // global loads overlap WMMA
        const v16bf af0 = load_frag(fA0 + cur * aStride, kb);
        const v16bf af1 = load_frag(fA1 + cur * aStride, kb);
        const v16bf bf0 = load_frag(fB0 + cur * bStride, kb);
        const v16bf bf1 = load_frag(fB1 + cur * bStride, kb);
        acc00 = bwmma(af0, bf0, acc00);
        acc01 = bwmma(af0, bf1, acc01);
        acc10 = bwmma(af1, bf0, acc10);
        acc11 = bwmma(af1, bf1, acc11);
        if (more) store_chunk(1 - cur);
        __syncthreads();
    }

    const int cn0 = n0 + wn * 32 + (lane & 15);
    const int cn1 = cn0 + 16;
    const int mb  = m0 + wm * 32 + (lane >> 4) * 8;
    const float bv0 = bias[cn0];
    const float bv1 = bias[cn1];
    #pragma unroll
    for (int r2 = 0; r2 < 8; ++r2) {
        const int g0 = mb + r2, g1 = mb + 16 + r2;
        if (g0 < M) {
            float v = acc00[r2] + bv0; if (act == 1) v = fmaxf(v, 0.0f); C[(size_t)g0 * N + cn0] = v;
            v       = acc01[r2] + bv1; if (act == 1) v = fmaxf(v, 0.0f); C[(size_t)g0 * N + cn1] = v;
        }
        if (g1 < M) {
            float v = acc10[r2] + bv0; if (act == 1) v = fmaxf(v, 0.0f); C[(size_t)g1 * N + cn0] = v;
            v       = acc11[r2] + bv1; if (act == 1) v = fmaxf(v, 0.0f); C[(size_t)g1 * N + cn1] = v;
        }
    }
}

// ---------- gemm_cat: A = [A0 | A1] (each [256,512]), K=1024, 64x64 tile, no edges ----------
__global__ __launch_bounds__(256) void gemm_cat(
    const float* __restrict__ A0, const float* __restrict__ A1,
    const float* __restrict__ B, float* __restrict__ C,
    const float* __restrict__ bias, int N, int act)
{
    __shared__ alignas(16) u16 sA[2][64 * LDS_PITCH];
    __shared__ alignas(16) u16 sB[2][64 * LDS_PITCH];

    const int tid  = threadIdx.x;
    const int lane = tid & 31;
    const int wave = tid >> 5;
    const int wm   = wave >> 2;   // 0..1
    const int wn   = wave & 3;    // 0..3
    const int m0   = blockIdx.y * 64;
    const int n0   = blockIdx.x * 64;

    const int srow = tid >> 2;
    const int skof = (tid & 3) * 8;
    const float* a0Row = A0 + (size_t)(m0 + srow) * 512 + skof;
    const float* a1Row = A1 + (size_t)(m0 + srow) * 512 + skof;
    const float* bRow  = B + (size_t)(n0 + srow) * 1024 + skof;

    v8f acc0 = {}, acc1 = {};
    float4 av0, av1, bv0, bv1;

    auto load_chunk = [&](int kc) {  // kc loop-uniform; split select is scalar
        const float* ar = (kc < 512) ? (a0Row + kc) : (a1Row + (kc - 512));
        const float4* ap = reinterpret_cast<const float4*>(ar);
        const float4* bp = reinterpret_cast<const float4*>(bRow + kc);
        av0 = ap[0]; av1 = ap[1]; bv0 = bp[0]; bv1 = bp[1];
    };

    load_chunk(0);
    st8bf(sA[0] + srow * LDS_PITCH + skof, av0, av1);
    st8bf(sB[0] + srow * LDS_PITCH + skof, bv0, bv1);
    __syncthreads();

    const int r  = lane & 15;
    const int kb = (lane >> 4) * 8;
    const size_t bufStride = 64 * LDS_PITCH;
    const u16* fB  = sB[0] + (wn * 16 + r) * LDS_PITCH;
    const u16* fA0 = sA[0] + (wm * 32 + r) * LDS_PITCH;
    const u16* fA1 = fA0 + 16 * LDS_PITCH;

    for (int i = 0; i < 32; ++i) {
        const int cur = i & 1;
        const bool more = (i + 1 < 32);
        if (more) load_chunk((i + 1) << 5);
        const v16bf bfrag  = load_frag(fB  + cur * bufStride, kb);
        const v16bf afrag0 = load_frag(fA0 + cur * bufStride, kb);
        const v16bf afrag1 = load_frag(fA1 + cur * bufStride, kb);
        acc0 = bwmma(afrag0, bfrag, acc0);
        acc1 = bwmma(afrag1, bfrag, acc1);
        if (more) {
            st8bf(sA[1 - cur] + srow * LDS_PITCH + skof, av0, av1);
            st8bf(sB[1 - cur] + srow * LDS_PITCH + skof, bv0, bv1);
        }
        __syncthreads();
    }

    const int cn = n0 + wn * 16 + (lane & 15);
    const int mb = m0 + wm * 32 + (lane >> 4) * 8;
    const float bv = bias[cn];
    #pragma unroll
    for (int r2 = 0; r2 < 8; ++r2) {
        float v = acc0[r2] + bv; if (act == 1) v = fmaxf(v, 0.0f);
        C[(size_t)(mb + r2) * N + cn] = v;
        v = acc1[r2] + bv; if (act == 1) v = fmaxf(v, 0.0f);
        C[(size_t)(mb + 16 + r2) * N + cn] = v;
    }
}

// ---------- gemm_dual: blockIdx.z selects operand set (gi / gh), K=512, no edges ----------
__global__ __launch_bounds__(256) void gemm_dual(
    const float* __restrict__ Ax, const float* __restrict__ Bx,
    float* __restrict__ Cx, const float* __restrict__ biasx,
    const float* __restrict__ Ay, const float* __restrict__ By,
    float* __restrict__ Cy, const float* __restrict__ biasy,
    int N, int K)
{
    const bool sec = (blockIdx.z != 0);
    const float* A = sec ? Ay : Ax;
    const float* B = sec ? By : Bx;
    float* C = sec ? Cy : Cx;
    const float* bias = sec ? biasy : biasx;

    __shared__ alignas(16) u16 sA[2][64 * LDS_PITCH];
    __shared__ alignas(16) u16 sB[2][64 * LDS_PITCH];

    const int tid  = threadIdx.x;
    const int lane = tid & 31;
    const int wave = tid >> 5;
    const int wm   = wave >> 2;
    const int wn   = wave & 3;
    const int m0   = blockIdx.y * 64;
    const int n0   = blockIdx.x * 64;

    const int srow = tid >> 2;
    const int skof = (tid & 3) * 8;
    const float* aRow = A + (size_t)(m0 + srow) * K + skof;
    const float* bRow = B + (size_t)(n0 + srow) * K + skof;

    v8f acc0 = {}, acc1 = {};
    float4 av0, av1, bv0, bv1;
    const int nk = K >> 5;

    auto load_chunk = [&](int kc) {
        const float4* ap = reinterpret_cast<const float4*>(aRow + kc);
        const float4* bp = reinterpret_cast<const float4*>(bRow + kc);
        av0 = ap[0]; av1 = ap[1]; bv0 = bp[0]; bv1 = bp[1];
    };

    load_chunk(0);
    st8bf(sA[0] + srow * LDS_PITCH + skof, av0, av1);
    st8bf(sB[0] + srow * LDS_PITCH + skof, bv0, bv1);
    __syncthreads();

    const int r  = lane & 15;
    const int kb = (lane >> 4) * 8;
    const size_t bufStride = 64 * LDS_PITCH;
    const u16* fB  = sB[0] + (wn * 16 + r) * LDS_PITCH;
    const u16* fA0 = sA[0] + (wm * 32 + r) * LDS_PITCH;
    const u16* fA1 = fA0 + 16 * LDS_PITCH;

    for (int i = 0; i < nk; ++i) {
        const int cur = i & 1;
        const bool more = (i + 1 < nk);
        if (more) load_chunk((i + 1) << 5);
        const v16bf bfrag  = load_frag(fB  + cur * bufStride, kb);
        const v16bf afrag0 = load_frag(fA0 + cur * bufStride, kb);
        const v16bf afrag1 = load_frag(fA1 + cur * bufStride, kb);
        acc0 = bwmma(afrag0, bfrag, acc0);
        acc1 = bwmma(afrag1, bfrag, acc1);
        if (more) {
            st8bf(sA[1 - cur] + srow * LDS_PITCH + skof, av0, av1);
            st8bf(sB[1 - cur] + srow * LDS_PITCH + skof, bv0, bv1);
        }
        __syncthreads();
    }

    const int cn = n0 + wn * 16 + (lane & 15);
    const int mb = m0 + wm * 32 + (lane >> 4) * 8;
    const float bv = bias[cn];
    #pragma unroll
    for (int r2 = 0; r2 < 8; ++r2) {
        C[(size_t)(mb + r2) * N + cn]      = acc0[r2] + bv;
        C[(size_t)(mb + 16 + r2) * N + cn] = acc1[r2] + bv;
    }
}

// ---------- conv weight repack: conv_w[32][2048][3][3] -> Wr[kk][oc][ic] bf16 ----------
__global__ void repack_convw(const float* __restrict__ w, u16* __restrict__ Wr) {
    int i = blockIdx.x * 256 + threadIdx.x;
    if (i >= 9 * 32 * 2048) return;
    int ic = i % 2048; int t = i / 2048; int oc = t % 32; int kk = t / 32;
    Wr[i] = f2bf(w[((size_t)oc * 2048 + ic) * 9 + kk]);
}

// ---------- implicit-GEMM conv: one image per block, Y[b][32][64] ----------
__global__ __launch_bounds__(256) void conv_wmma(
    const float* __restrict__ X,     // [B,2048,8,8]
    const u16*   __restrict__ Wr,    // [9][32][2048] bf16
    const float* __restrict__ bias,  // [32]
    float* __restrict__ Y)           // [B,32,64]
{
    __shared__ alignas(16) u16 sW[9][32 * LDS_PITCH];
    __shared__ alignas(16) u16 sX[65 * LDS_PITCH];     // row 64 = zeros (padding)

    const int b    = blockIdx.x;
    const int tid  = threadIdx.x;
    const int lane = tid & 31;
    const int wave = tid >> 5;
    const int wm   = wave >> 2;
    const int wn   = wave & 3;
    const float* Xb = X + (size_t)b * 2048 * 64;

    if (tid < LDS_PITCH) sX[64 * LDS_PITCH + tid] = 0;

    v8f acc = {};
    const int r  = lane & 15;
    const int kb = (lane >> 4) * 8;
    const int p  = wn * 16 + r;
    const int py = p >> 3, px = p & 7;

    for (int ic0 = 0; ic0 < 2048; ic0 += 32) {
        {   // stage activations transposed: sX[pix][j] = bf16(Xb[(ic0+j)*64 + pix])
            const int j = tid & 31;
            const int y = tid >> 5;
            const float4* xp = reinterpret_cast<const float4*>(Xb + (size_t)(ic0 + j) * 64 + y * 8);
            const float4 r0 = xp[0], r1 = xp[1];
            u16* dst = sX + (y * 8) * LDS_PITCH + j;
            dst[0 * LDS_PITCH] = f2bf(r0.x); dst[1 * LDS_PITCH] = f2bf(r0.y);
            dst[2 * LDS_PITCH] = f2bf(r0.z); dst[3 * LDS_PITCH] = f2bf(r0.w);
            dst[4 * LDS_PITCH] = f2bf(r1.x); dst[5 * LDS_PITCH] = f2bf(r1.y);
            dst[6 * LDS_PITCH] = f2bf(r1.z); dst[7 * LDS_PITCH] = f2bf(r1.w);
        }
        {   // stage weights for all 9 taps
            const int row = tid >> 3;
            const int kof = (tid & 7) * 4;
            #pragma unroll
            for (int kk = 0; kk < 9; ++kk) {
                const u16* wp = Wr + ((size_t)kk * 32 + row) * 2048 + ic0 + kof;
                *reinterpret_cast<uint2*>(sW[kk] + row * LDS_PITCH + kof) =
                    *reinterpret_cast<const uint2*>(wp);
            }
        }
        __syncthreads();

        #pragma unroll
        for (int kk = 0; kk < 9; ++kk) {
            const int dy = kk / 3 - 1, dx = kk % 3 - 1;
            const int sy = py + dy, sx = px + dx;
            const int prow = (sy >= 0 && sy < 8 && sx >= 0 && sx < 8) ? (sy * 8 + sx) : 64;
            const v16bf bfrag = load_frag(sX + prow * LDS_PITCH, kb);
            const v16bf afrag = load_frag(sW[kk] + (wm * 16 + r) * LDS_PITCH, kb);
            acc = bwmma(afrag, bfrag, acc);
        }
        __syncthreads();
    }

    const int oc = wm * 16 + (lane >> 4) * 8;
    #pragma unroll
    for (int r2 = 0; r2 < 8; ++r2)
        Y[(size_t)b * 2048 + (oc + r2) * 64 + p] = acc[r2] + bias[oc + r2];
}

// ---------- fused BatchNorm (stats + apply), one column per block ----------
__global__ __launch_bounds__(256) void bn_fused(const float* __restrict__ f,
                                                const float* __restrict__ g,
                                                const float* __restrict__ bb,
                                                float* __restrict__ out) {
    __shared__ float s1[256], s2[256];
    const int h = blockIdx.x, t = threadIdx.x;
    const float v = f[t * 512 + h];
    s1[t] = v; s2[t] = v * v; __syncthreads();
    for (int o = 128; o > 0; o >>= 1) {
        if (t < o) { s1[t] += s1[t + o]; s2[t] += s2[t + o]; }
        __syncthreads();
    }
    const float mu  = s1[0] * (1.0f / 256.0f);
    const float var = s2[0] * (1.0f / 256.0f) - mu * mu;
    const float rv  = rsqrtf(var + 1e-5f);
    out[t * 512 + h] = (v - mu) * rv * g[h] + bb[h];
}

// ---------- unpack packed captions -> xs[t][b][e] with zeros on pad ----------
__global__ void unpack_emb(const float* __restrict__ emb, const int* __restrict__ pack_idx,
                           float* __restrict__ xs) {
    const int i = blockIdx.x * 256 + threadIdx.x;
    if (i >= 48 * 256 * 512) return;
    const int e = i & 511; const int bt = i >> 9;
    const int b = bt & 255; const int t = bt >> 8;
    const int idx = pack_idx[b * 48 + t];
    xs[i] = (idx >= 0) ? emb[(size_t)idx * 512 + e] : 0.0f;
}

// ---------- softmax over 512 + gating; writes applied[256,512] ----------
__global__ __launch_bounds__(256) void softmax_apply(const float* __restrict__ attn,
                                                     const float* __restrict__ fbn,
                                                     float* __restrict__ applied) {
    __shared__ float red[256];
    const int row = blockIdx.x, t = threadIdx.x;
    const float* a = attn + row * 512;
    const float v0 = a[t], v1 = a[t + 256];
    red[t] = fmaxf(v0, v1); __syncthreads();
    for (int o = 128; o > 0; o >>= 1) { if (t < o) red[t] = fmaxf(red[t], red[t + o]); __syncthreads(); }
    const float mx = red[0]; __syncthreads();
    const float e0 = __expf(v0 - mx), e1 = __expf(v1 - mx);
    red[t] = e0 + e1; __syncthreads();
    for (int o = 128; o > 0; o >>= 1) { if (t < o) red[t] += red[t + o]; __syncthreads(); }
    const float inv = 1.0f / red[0];
    applied[row * 512 + t]       = fbn[row * 512 + t]       * e0 * inv;
    applied[row * 512 + t + 256] = fbn[row * 512 + t + 256] * e1 * inv;
}

// ---------- GRU gate math; captures hiddens at t == length-1 ----------
__global__ void gru_gate(const float* __restrict__ gi, const float* __restrict__ gh,
                         float* __restrict__ hx, const int* __restrict__ lengths,
                         int t, float* __restrict__ hiddens) {
    const int i = blockIdx.x * 256 + threadIdx.x;
    if (i >= 256 * 512) return;
    const int row = i >> 9, h = i & 511;
    const float* gir = gi + (size_t)row * 1536;
    const float* ghr = gh + (size_t)row * 1536;
    const float ir = gir[h], iz = gir[512 + h], in_ = gir[1024 + h];
    const float hr = ghr[h], hz = ghr[512 + h], hn  = ghr[1024 + h];
    const float r = 1.0f / (1.0f + __expf(-(ir + hr)));
    const float z = 1.0f / (1.0f + __expf(-(iz + hz)));
    const float n = tanhf(in_ + r * hn);
    const float hnew = (1.0f - z) * n + z * hx[i];
    hx[i] = hnew;
    if (lengths[row] - 1 == t) hiddens[i] = hnew;
}

// ---------- final concat + dot + sigmoid ----------
__global__ void build_xfin(const float* __restrict__ hiddens, const float* __restrict__ fbn,
                           float* __restrict__ xfin) {
    const int i = blockIdx.x * 256 + threadIdx.x;
    const int c = i & 1023; const int row = i >> 10;
    xfin[i] = (c < 512) ? hiddens[row * 512 + c] : fbn[row * 512 + (c - 512)];
}

__global__ __launch_bounds__(256) void lin3_sigmoid(const float* __restrict__ xl,
                                                    const float* __restrict__ w,
                                                    const float* __restrict__ b,
                                                    float* __restrict__ out) {
    __shared__ float red[256];
    const int row = blockIdx.x, t = threadIdx.x;
    const float* x = xl + (size_t)row * 1024;
    red[t] = x[t] * w[t] + x[t + 256] * w[t + 256] + x[t + 512] * w[t + 512] + x[t + 768] * w[t + 768];
    __syncthreads();
    for (int o = 128; o > 0; o >>= 1) { if (t < o) red[t] += red[t + o]; __syncthreads(); }
    if (t == 0) out[row] = 1.0f / (1.0f + __expf(-(red[0] + b[0])));
}

// ---------- workspace layout (float units) ----------
static const size_t OFF_CONVOUT = 0;                          // 256*2048
static const size_t OFF_FBUF    = OFF_CONVOUT + 524288;       // 256*512
static const size_t OFF_FBN     = OFF_FBUF + 131072;          // 256*512
static const size_t OFF_EMB     = OFF_FBN + 131072;           // 9436*512
static const size_t OFF_XS      = OFF_EMB + 4831232;          // 48*256*512
static const size_t OFF_APPL    = OFF_XS + 6291456;           // 256*512
static const size_t OFF_ATTN    = OFF_APPL + 131072;          // 256*512
static const size_t OFF_INP     = OFF_ATTN + 131072;          // 256*512
static const size_t OFF_GI      = OFF_INP + 131072;           // 256*1536
static const size_t OFF_GH      = OFF_GI + 393216;            // 256*1536
static const size_t OFF_HX      = OFF_GH + 393216;            // 256*512
static const size_t OFF_XFIN    = OFF_HX + 131072;            // 256*1024
static const size_t OFF_XL      = OFF_XFIN + 262144;          // 256*1024
static const size_t OFF_WR      = OFF_XL + 262144;            // 589824 u16 = 294912 f

extern "C" void kernel_launch(void* const* d_in, const int* in_sizes, int n_in,
                              void* d_out, int out_size, void* d_ws, size_t ws_size,
                              hipStream_t stream) {
    const float* features = (const float*)d_in[0];
    const float* captions = (const float*)d_in[1];
    const float* states   = (const float*)d_in[2];
    const int*   pack_idx = (const int*)d_in[3];
    const int*   lengths  = (const int*)d_in[4];
    const float* conv_w   = (const float*)d_in[5];
    const float* conv_b   = (const float*)d_in[6];
    const float* fc_w     = (const float*)d_in[7];
    const float* fc_b     = (const float*)d_in[8];
    const float* bn_g     = (const float*)d_in[9];
    const float* bn_b     = (const float*)d_in[10];
    const float* fc2_w    = (const float*)d_in[11];
    const float* fc2_b    = (const float*)d_in[12];
    const float* attn_w   = (const float*)d_in[13];
    const float* attn_b   = (const float*)d_in[14];
    const float* comb_w   = (const float*)d_in[15];
    const float* comb_b   = (const float*)d_in[16];
    const float* gru_wih  = (const float*)d_in[17];
    const float* gru_whh  = (const float*)d_in[18];
    const float* gru_bih  = (const float*)d_in[19];
    const float* gru_bhh  = (const float*)d_in[20];
    const float* lin_w    = (const float*)d_in[21];
    const float* lin_b    = (const float*)d_in[22];
    const float* lin3_w   = (const float*)d_in[23];
    const float* lin3_b   = (const float*)d_in[24];
    (void)in_sizes; (void)n_in; (void)out_size; (void)ws_size;

    float* ws = (float*)d_ws;
    float* out_sig = (float*)d_out;    // [256]
    float* out_hid = out_sig + 256;    // [256,512]

    // 1) conv weight repack + implicit-GEMM conv (WMMA bf16)
    repack_convw<<<(9 * 32 * 2048 + 255) / 256, 256, 0, stream>>>(conv_w, (u16*)(ws + OFF_WR));
    conv_wmma<<<256, 256, 0, stream>>>(features, (const u16*)(ws + OFF_WR), conv_b, ws + OFF_CONVOUT);

    // 2) fc + fused BatchNorm
    gemm_big<<<dim3(8, 2), 256, 0, stream>>>(ws + OFF_CONVOUT, fc_w, ws + OFF_FBUF, fc_b,
                                             256, 512, 2048, 0);
    bn_fused<<<512, 256, 0, stream>>>(ws + OFF_FBUF, bn_g, bn_b, ws + OFF_FBN);

    // 3) fc2 on packed captions (dominant GEMM, HBM-bound)
    gemm_big<<<dim3(8, 74), 256, 0, stream>>>(captions, fc2_w, ws + OFF_EMB, fc2_b,
                                              9436, 512, 10000, 0);
    unpack_emb<<<(48 * 256 * 512) / 256, 256, 0, stream>>>(ws + OFF_EMB, pack_idx, ws + OFF_XS);

    // 4) hx <- states[0]
    hipMemcpyAsync(ws + OFF_HX, states, 256 * 512 * sizeof(float),
                   hipMemcpyDeviceToDevice, stream);

    // 5) 48-step attention GRU (5 launches/step)
    for (int t = 0; t < 48; ++t) {
        const float* xs_t = ws + OFF_XS + (size_t)t * 256 * 512;
        gemm_cat<<<dim3(8, 4), 256, 0, stream>>>(xs_t, ws + OFF_HX, attn_w, ws + OFF_ATTN,
                                                 attn_b, 512, 0);
        softmax_apply<<<256, 256, 0, stream>>>(ws + OFF_ATTN, ws + OFF_FBN, ws + OFF_APPL);
        gemm_cat<<<dim3(8, 4), 256, 0, stream>>>(xs_t, ws + OFF_APPL, comb_w, ws + OFF_INP,
                                                 comb_b, 512, 1);
        gemm_dual<<<dim3(24, 4, 2), 256, 0, stream>>>(
            ws + OFF_INP, gru_wih, ws + OFF_GI, gru_bih,
            ws + OFF_HX,  gru_whh, ws + OFF_GH, gru_bhh, 1536, 512);
        gru_gate<<<(256 * 512) / 256, 256, 0, stream>>>(ws + OFF_GI, ws + OFF_GH, ws + OFF_HX,
                                                        lengths, t, out_hid);
    }

    // 6) classifier head
    build_xfin<<<(256 * 1024) / 256, 256, 0, stream>>>(out_hid, ws + OFF_FBN, ws + OFF_XFIN);
    gemm_big<<<dim3(16, 2), 256, 0, stream>>>(ws + OFF_XFIN, lin_w, ws + OFF_XL, lin_b,
                                              256, 1024, 1024, 1);
    lin3_sigmoid<<<256, 256, 0, stream>>>(ws + OFF_XL, lin3_w, lin3_b, out_sig);
}
